// WindowAttention_21019569946845
// MI455X (gfx1250) — compile-verified, matrix operand order checked by reading, not run on hardware
//
#include <hip/hip_runtime.h>

// ---------------- CDNA5 (gfx1250) wave32 WMMA types ----------------
typedef _Float16 half16_t __attribute__((ext_vector_type(16)));
typedef _Float16 half8_t  __attribute__((ext_vector_type(8)));
typedef float    f32x8    __attribute__((ext_vector_type(8)));

union AFrag { half16_t v; half8_t h8[2]; _Float16 h[16]; };

#define BATCH 8
#define HH 224
#define WW2 224
#define NPIX (BATCH*HH*WW2)   // 401408
#define WSZ 7
#define DISP 3
#define NWH 32                 // windows per side
#define HEADS 3
#define HD 32
#define DIM 96
#define SCALE 0.17677669529663687f

// workspace byte offsets (all 256B aligned)
#define QKV_OFF   0ul
#define T1_OFF    231211008ul
#define T2_OFF    (231211008ul + 51380224ul)
#define AO_OFF    (231211008ul + 51380224ul + 25690112ul)

// ---------------------------------------------------------------------------
// Kernel 1: conv1 (3 -> 64), direct, fp32 compute, ReLU, store NHWC f16
// ---------------------------------------------------------------------------
__global__ void conv1_kernel(const float* __restrict__ x,
                             const float* __restrict__ w,
                             const float* __restrict__ bias,
                             _Float16* __restrict__ t1) {
  __shared__ float sw[64*27];
  __shared__ float sb[64];
  for (int i = threadIdx.x; i < 64*27; i += blockDim.x) sw[i] = w[i];
  for (int i = threadIdx.x; i < 64;    i += blockDim.x) sb[i] = bias[i];
  __syncthreads();
  long gid = (long)blockIdx.x * blockDim.x + threadIdx.x;   // pixel*64 + co
  int co = (int)(gid & 63);
  long pix = gid >> 6;
  int wc = (int)(pix % WW2);
  int hr = (int)((pix / WW2) % HH);
  int b  = (int)(pix / (WW2*HH));
  float acc = sb[co];
  const float* wp = &sw[co*27];
  for (int c = 0; c < 3; ++c)
    for (int dy = 0; dy < 3; ++dy) {
      int hy = hr + dy - 1;
      if (hy < 0 || hy >= HH) continue;
      const float* row = x + (((long)b*3 + c)*HH + hy)*WW2;
      for (int dx = 0; dx < 3; ++dx) {
        int wx = wc + dx - 1;
        if (wx < 0 || wx >= WW2) continue;
        acc += row[wx] * wp[(c*3+dy)*3+dx];
      }
    }
  acc = acc > 0.f ? acc : 0.f;
  t1[pix*64 + co] = (_Float16)acc;
}

// ---------------------------------------------------------------------------
// Kernel 2: conv2 (64 -> 32) implicit GEMM via WMMA. 4 waves/block, each wave
// computes a 16-pixel x 32-cout tile. Weights pre-packed to B-fragment order.
// ---------------------------------------------------------------------------
__global__ void __launch_bounds__(128) conv2_kernel(const _Float16* __restrict__ t1,
                                                    const float* __restrict__ w,
                                                    const float* __restrict__ bias,
                                                    _Float16* __restrict__ t2) {
  // frag f = (tap*2 + kc)*2 + nt  : [36][32 lanes][16 halves]
  __shared__ __attribute__((aligned(64))) _Float16 sB[36*512];
  __shared__ float sb[32];
  for (int e = threadIdx.x; e < 36*512; e += 128) {
    int f = e >> 9, l = (e >> 4) & 31, i = e & 15;
    int tap = f >> 2, kc = (f >> 1) & 1, nt = f & 1;
    int kb = (l >> 4) * 8;
    int k  = kc*32 + kb + (i & 7) + ((i >> 3) << 4);      // cin
    int n  = nt*16 + (l & 15);                            // cout
    int dy = tap/3, dx = tap%3;
    sB[e] = (_Float16)w[((n*64 + k)*3 + dy)*3 + dx];
  }
  if (threadIdx.x < 32) sb[threadIdx.x] = bias[threadIdx.x];
  __syncthreads();

  int lane = threadIdx.x & 31, wv = threadIdx.x >> 5;
  int tile = blockIdx.x*4 + wv;              // 25088 tiles of 16 pixels
  int wseg = tile % 14;
  int hr   = (tile/14) % HH;
  int b    = tile / (14*HH);
  int w0   = wseg*16;
  int mrow = lane & 15;
  int kb   = (lane >> 4) * 8;

  f32x8 acc0 = {}, acc1 = {};
  for (int tap = 0; tap < 9; ++tap) {
    int dy = tap/3, dx = tap%3;
    int hy = hr + dy - 1;
    if (hy < 0 || hy >= HH) continue;                     // uniform per wave
    int wx = w0 + mrow + dx - 1;
    bool inb = (wx >= 0) && (wx < WW2);
    const _Float16* src = t1 + ((long)(b*HH + hy)*WW2 + wx)*64;
    if (inb) __builtin_prefetch(src, 0, 1);
    for (int kc = 0; kc < 2; ++kc) {
      AFrag a;
      if (inb) {
        a.h8[0] = *(const half8_t*)(src + kc*32 + kb);
        a.h8[1] = *(const half8_t*)(src + kc*32 + kb + 16);
      } else {
        half8_t z = {}; a.h8[0] = z; a.h8[1] = z;
      }
      half16_t b0 = *(const half16_t*)&sB[(((tap*2+kc)*2+0)<<9) + lane*16];
      half16_t b1 = *(const half16_t*)&sB[(((tap*2+kc)*2+1)<<9) + lane*16];
      acc0 = __builtin_amdgcn_wmma_f32_16x16x32_f16(false, a.v, false, b0, (short)0, acc0, false, false);
      acc1 = __builtin_amdgcn_wmma_f32_16x16x32_f16(false, a.v, false, b1, (short)0, acc1, false, false);
    }
  }
  int g = lane >> 4, n0 = lane & 15;
  long rowbase = (long)(b*HH + hr)*WW2 + w0;
  for (int r = 0; r < 8; ++r) {
    long pix = rowbase + r + 8*g;
    float v0 = acc0[r] + sb[n0];
    float v1 = acc1[r] + sb[16+n0];
    v0 = v0 > 0.f ? v0 : 0.f;  v1 = v1 > 0.f ? v1 : 0.f;
    _Float16* dst = t2 + pix*32;
    dst[n0]      = (_Float16)v0;
    dst[16 + n0] = (_Float16)v1;
  }
}

// ---------------------------------------------------------------------------
// Kernel 3: conv3 (32 -> 96) implicit GEMM via WMMA, writes into qkv[.., pofs+..]
// ---------------------------------------------------------------------------
__global__ void __launch_bounds__(128) conv3_kernel(const _Float16* __restrict__ t2,
                                                    const float* __restrict__ w,
                                                    const float* __restrict__ bias,
                                                    _Float16* __restrict__ qkv,
                                                    int pofs) {
  // frag f = tap*6 + nt : [54][32][16]
  __shared__ __attribute__((aligned(64))) _Float16 sB[54*512];
  __shared__ float sb[96];
  for (int e = threadIdx.x; e < 54*512; e += 128) {
    int f = e >> 9, l = (e >> 4) & 31, i = e & 15;
    int tap = f/6, nt = f%6;
    int kb = (l >> 4) * 8;
    int k  = kb + (i & 7) + ((i >> 3) << 4);              // cin (0..31)
    int n  = nt*16 + (l & 15);                            // cout (0..95)
    int dy = tap/3, dx = tap%3;
    sB[e] = (_Float16)w[((n*32 + k)*3 + dy)*3 + dx];
  }
  if (threadIdx.x < 96) sb[threadIdx.x] = bias[threadIdx.x];
  __syncthreads();

  int lane = threadIdx.x & 31, wv = threadIdx.x >> 5;
  int tile = blockIdx.x*4 + wv;
  int wseg = tile % 14;
  int hr   = (tile/14) % HH;
  int b    = tile / (14*HH);
  int w0   = wseg*16;
  int mrow = lane & 15;
  int kb   = (lane >> 4) * 8;

  f32x8 acc[6] = {};
  for (int tap = 0; tap < 9; ++tap) {
    int dy = tap/3, dx = tap%3;
    int hy = hr + dy - 1;
    if (hy < 0 || hy >= HH) continue;
    int wx = w0 + mrow + dx - 1;
    bool inb = (wx >= 0) && (wx < WW2);
    const _Float16* src = t2 + ((long)(b*HH + hy)*WW2 + wx)*32;
    AFrag a;
    if (inb) {
      a.h8[0] = *(const half8_t*)(src + kb);
      a.h8[1] = *(const half8_t*)(src + kb + 16);
    } else {
      half8_t z = {}; a.h8[0] = z; a.h8[1] = z;
    }
    for (int nt = 0; nt < 6; ++nt) {
      half16_t bf = *(const half16_t*)&sB[((tap*6+nt)<<9) + lane*16];
      acc[nt] = __builtin_amdgcn_wmma_f32_16x16x32_f16(false, a.v, false, bf, (short)0, acc[nt], false, false);
    }
  }
  int g = lane >> 4, n0 = lane & 15;
  long rowbase = (long)(b*HH + hr)*WW2 + w0;
  for (int r = 0; r < 8; ++r) {
    long pix = rowbase + r + 8*g;
    _Float16* dst = qkv + pix*288 + pofs;
    for (int nt = 0; nt < 6; ++nt)
      dst[nt*16 + n0] = (_Float16)(acc[nt][r] + sb[nt*16 + n0]);
  }
}

// ---------------------------------------------------------------------------
// Kernel 4: shifted-window attention. One block (4 waves) per (batch, window),
// loop over 3 heads. Q/K staged into LDS with gfx1250 async global->LDS DMA
// (ASYNCcnt), V transposed via ds stores. QK^T and PV via WMMA (49 tokens
// padded to 64), softmax in registers with wave32 shfl_xor reductions.
// Cyclic shift folded into gather/scatter indices.
// ---------------------------------------------------------------------------
__global__ void __launch_bounds__(128) attn_kernel(const _Float16* __restrict__ qkv,
                                                   const float* __restrict__ pos_emb,
                                                   _Float16* __restrict__ ao) {
  __shared__ __attribute__((aligned(64))) _Float16 Ql[64*32];
  __shared__ __attribute__((aligned(64))) _Float16 Kl[64*32];
  __shared__ __attribute__((aligned(64))) _Float16 Vt[32*64];   // transposed: [ch][tok]
  __shared__ __attribute__((aligned(64))) _Float16 Pl[4*16*64]; // per-wave P tiles
  __shared__ float pe[169];

  int wid = blockIdx.x & 1023;
  int b   = blockIdx.x >> 10;
  int wh  = wid >> 5, ww = wid & 31;
  for (int i = threadIdx.x; i < 169; i += 128) pe[i] = pos_emb[i];
  int lane = threadIdx.x & 31, wv = threadIdx.x >> 5;
  int kb = (lane >> 4) * 8, g = lane >> 4, jcol = lane & 15;
  bool ul = (wh == NWH-1), lr = (ww == NWH-1);

  for (int head = 0; head < HEADS; ++head) {
    __syncthreads();
    if (threadIdx.x < 64) {
      int tok = threadIdx.x;
      if (tok < 49) {
        int ty = tok/7, tx = tok%7;
        int hs = (wh*7 + ty + DISP) % HH;
        int ws = (ww*7 + tx + DISP) % WW2;
        const _Float16* srcq = qkv + ((long)(b*HH + hs)*WW2 + ws)*288 + head*32;
        const _Float16* srck = srcq + 96;
        const _Float16* srcv = srcq + 192;
        unsigned ldq = (unsigned)(size_t)&Ql[tok*32];   // LDS byte offset (addr[31:0])
        unsigned ldk = (unsigned)(size_t)&Kl[tok*32];
        // 64B per token per matrix: 4 x b128 async DMA, global->LDS, ASYNCcnt
        asm volatile(
          "global_load_async_to_lds_b128 %0, %2, off\n\t"
          "global_load_async_to_lds_b128 %0, %2, off offset:16\n\t"
          "global_load_async_to_lds_b128 %0, %2, off offset:32\n\t"
          "global_load_async_to_lds_b128 %0, %2, off offset:48\n\t"
          "global_load_async_to_lds_b128 %1, %3, off\n\t"
          "global_load_async_to_lds_b128 %1, %3, off offset:16\n\t"
          "global_load_async_to_lds_b128 %1, %3, off offset:32\n\t"
          "global_load_async_to_lds_b128 %1, %3, off offset:48"
          :: "v"(ldq), "v"(ldk), "v"(srcq), "v"(srck) : "memory");
        for (int i = 0; i < 32; ++i) Vt[i*64 + tok] = srcv[i];   // transpose scatter
      } else {
        half8_t z = {};
        for (int i = 0; i < 32; i += 8) { *(half8_t*)&Ql[tok*32+i] = z; *(half8_t*)&Kl[tok*32+i] = z; }
        for (int i = 0; i < 32; ++i) Vt[i*64 + tok] = (_Float16)0.f;
      }
    }
    asm volatile("s_wait_asynccnt 0x0" ::: "memory");   // async DMA fence
    __syncthreads();

    // ---- dots = Q @ K^T : wave wv owns M-tile wv (16 query tokens) ----
    int mrow = wv*16 + (lane & 15);
    AFrag qa;
    qa.h8[0] = *(const half8_t*)&Ql[mrow*32 + kb];
    qa.h8[1] = *(const half8_t*)&Ql[mrow*32 + kb + 16];
    f32x8 dots[4];
    for (int nt = 0; nt < 4; ++nt) {
      int ncol = nt*16 + (lane & 15);
      AFrag kf;
      kf.h8[0] = *(const half8_t*)&Kl[ncol*32 + kb];
      kf.h8[1] = *(const half8_t*)&Kl[ncol*32 + kb + 16];
      f32x8 c = {};
      dots[nt] = __builtin_amdgcn_wmma_f32_16x16x32_f16(false, qa.v, false, kf.v, (short)0, c, false, false);
    }

    // ---- scale + rel-pos bias + shift masks + softmax (row = fixed r,g) ----
    for (int r = 0; r < 8; ++r) {
      int i = wv*16 + r + 8*g;            // query token index (padded to 64)
      int tyi = i/7, txi = i%7;
      float rowv[4];
      float rmax = -3.0e38f;
      for (int nt = 0; nt < 4; ++nt) {
        int j = nt*16 + jcol;
        float v;
        if (i < 49 && j < 49) {
          int tyj = j/7, txj = j%7;
          v = dots[nt][r] * SCALE + pe[(tyj - tyi + 6)*13 + (txj - txi + 6)];
          if (ul && (((tyi >= 4) ? 1 : 0) ^ ((tyj >= 4) ? 1 : 0))) v = -1.0e30f;
          if (lr && (((txi >= 4) ? 1 : 0) ^ ((txj >= 4) ? 1 : 0))) v = -1.0e30f;
        } else v = -1.0e30f;
        rowv[nt] = v;
        rmax = fmaxf(rmax, v);
      }
      for (int m = 1; m < 16; m <<= 1) rmax = fmaxf(rmax, __shfl_xor(rmax, m, 32));
      float rsum = 0.f;
      for (int nt = 0; nt < 4; ++nt) { rowv[nt] = __expf(rowv[nt] - rmax); rsum += rowv[nt]; }
      for (int m = 1; m < 16; m <<= 1) rsum += __shfl_xor(rsum, m, 32);
      float inv = 1.0f / rsum;
      int prow = r + 8*g;
      for (int nt = 0; nt < 4; ++nt)
        Pl[(wv*16 + prow)*64 + nt*16 + jcol] = (_Float16)(rowv[nt] * inv);
    }

    // ---- out = P @ V : K = 64 tokens (2 chunks), N = 32 channels (2 tiles) ----
    f32x8 oacc0 = {}, oacc1 = {};
    int prw = lane & 15;
    for (int kc = 0; kc < 2; ++kc) {
      AFrag pa;
      pa.h8[0] = *(const half8_t*)&Pl[(wv*16 + prw)*64 + kc*32 + kb];
      pa.h8[1] = *(const half8_t*)&Pl[(wv*16 + prw)*64 + kc*32 + kb + 16];
      AFrag v0, v1;
      int n0 = (lane & 15), n1 = 16 + (lane & 15);
      v0.h8[0] = *(const half8_t*)&Vt[n0*64 + kc*32 + kb];
      v0.h8[1] = *(const half8_t*)&Vt[n0*64 + kc*32 + kb + 16];
      v1.h8[0] = *(const half8_t*)&Vt[n1*64 + kc*32 + kb];
      v1.h8[1] = *(const half8_t*)&Vt[n1*64 + kc*32 + kb + 16];
      oacc0 = __builtin_amdgcn_wmma_f32_16x16x32_f16(false, pa.v, false, v0.v, (short)0, oacc0, false, false);
      oacc1 = __builtin_amdgcn_wmma_f32_16x16x32_f16(false, pa.v, false, v1.v, (short)0, oacc1, false, false);
    }

    // ---- scatter back to original (un-shifted) coords ----
    for (int r = 0; r < 8; ++r) {
      int i = wv*16 + r + 8*g;
      if (i < 49) {
        int ty = i/7, tx = i%7;
        int hd = (wh*7 + ty + DISP) % HH;
        int wd = (ww*7 + tx + DISP) % WW2;
        _Float16* dst = ao + ((long)(b*HH + hd)*WW2 + wd)*96 + head*32;
        dst[jcol]      = (_Float16)oacc0[r];
        dst[16 + jcol] = (_Float16)oacc1[r];
      }
    }
  }
}

// ---------------------------------------------------------------------------
// Kernel 5: output projection  out = ao @ out_w + out_b  (M=NPIX, N=K=96)
// ---------------------------------------------------------------------------
__global__ void __launch_bounds__(128) proj_kernel(const _Float16* __restrict__ ao,
                                                   const float* __restrict__ outw,
                                                   const float* __restrict__ outb,
                                                   float* __restrict__ out) {
  // frag f = kc*6 + nt : [18][32][16]
  __shared__ __attribute__((aligned(64))) _Float16 sB[18*512];
  __shared__ float sb[96];
  for (int e = threadIdx.x; e < 18*512; e += 128) {
    int f = e >> 9, l = (e >> 4) & 31, i = e & 15;
    int kc = f/6, nt = f%6;
    int kbb = (l >> 4) * 8;
    int k = kc*32 + kbb + (i & 7) + ((i >> 3) << 4);
    int n = nt*16 + (l & 15);
    sB[e] = (_Float16)outw[k*96 + n];
  }
  if (threadIdx.x < 96) sb[threadIdx.x] = outb[threadIdx.x];
  __syncthreads();

  int lane = threadIdx.x & 31, wv = threadIdx.x >> 5;
  long pix0 = ((long)blockIdx.x*4 + wv) * 16;
  int kb = (lane >> 4) * 8;
  const _Float16* src = ao + (pix0 + (lane & 15))*96;
  __builtin_prefetch(src, 0, 1);

  f32x8 acc[6] = {};
  for (int kc = 0; kc < 3; ++kc) {
    AFrag a;
    a.h8[0] = *(const half8_t*)(src + kc*32 + kb);
    a.h8[1] = *(const half8_t*)(src + kc*32 + kb + 16);
    for (int nt = 0; nt < 6; ++nt) {
      half16_t bf = *(const half16_t*)&sB[((kc*6+nt)<<9) + lane*16];
      acc[nt] = __builtin_amdgcn_wmma_f32_16x16x32_f16(false, a.v, false, bf, (short)0, acc[nt], false, false);
    }
  }
  int g = lane >> 4, n0 = lane & 15;
  for (int r = 0; r < 8; ++r) {
    float* dst = out + (pix0 + r + 8*g)*96;
    for (int nt = 0; nt < 6; ++nt)
      dst[nt*16 + n0] = acc[nt][r] + sb[nt*16 + n0];
  }
}

// ---------------------------------------------------------------------------
extern "C" void kernel_launch(void* const* d_in, const int* in_sizes, int n_in,
                              void* d_out, int out_size, void* d_ws, size_t ws_size,
                              hipStream_t stream) {
  (void)in_sizes; (void)n_in; (void)out_size; (void)ws_size;
  const float* x = (const float*)d_in[0];
  char* ws = (char*)d_ws;
  _Float16* qkv = (_Float16*)(ws + QKV_OFF);   // [8,224,224,288] f16
  _Float16* t1  = (_Float16*)(ws + T1_OFF);    // [8,224,224,64]  f16 (reused per path)
  _Float16* t2  = (_Float16*)(ws + T2_OFF);    // [8,224,224,32]  f16 (reused per path)
  _Float16* ao  = (_Float16*)(ws + AO_OFF);    // [8,224,224,96]  f16

  for (int p = 0; p < 3; ++p) {                // p=0 -> q, 1 -> k, 2 -> v
    const float* w1 = (const float*)d_in[1 + 6*p];
    const float* b1 = (const float*)d_in[2 + 6*p];
    const float* w2 = (const float*)d_in[3 + 6*p];
    const float* b2 = (const float*)d_in[4 + 6*p];
    const float* w3 = (const float*)d_in[5 + 6*p];
    const float* b3 = (const float*)d_in[6 + 6*p];
    conv1_kernel<<<NPIX*64/256, 256, 0, stream>>>(x, w1, b1, t1);
    conv2_kernel<<<6272, 128, 0, stream>>>(t1, w2, b2, t2);
    conv3_kernel<<<6272, 128, 0, stream>>>(t2, w3, b3, qkv, p*96);
  }
  attn_kernel<<<BATCH*1024, 128, 0, stream>>>(qkv, (const float*)d_in[19], ao);
  proj_kernel<<<6272, 128, 0, stream>>>(ao, (const float*)d_in[20],
                                        (const float*)d_in[21], (float*)d_out);
}
